// GatedLinearAttention_19825569038706
// MI455X (gfx1250) — matrix operand order repaired and checked by reading, not compiled, not run on hardware
//
#include <hip/hip_runtime.h>

// ---------------------------------------------------------------------------
// Gated linear attention for MI455X (gfx1250): all GEMM-shaped math on
// v_wmma_f32_16x16x32_bf16 (wave32 WMMA, f32 accumulate).  The dense GEMMs
// stage their shared B tile through LDS with GLOBAL_LOAD_ASYNC_TO_LDS_B128
// (ASYNCcnt), double-buffered so the DMA overlaps the WMMAs.
// ---------------------------------------------------------------------------

#define Bsz 4
#define Tsz 1024
#define Msz 512
#define Hsz 8
#define Dk 64
#define Dv 64
#define HK 512   // H*K == H*V == M here
#define EPSv 1e-6f

typedef __attribute__((ext_vector_type(8)))  float        v8f;
typedef __attribute__((ext_vector_type(16))) __bf16       v16bf;
typedef __attribute__((ext_vector_type(8)))  unsigned int v8u;

static __device__ __forceinline__ unsigned short f2bf(float f) {
  unsigned int u = __float_as_uint(f);
  u = (u + 0x7FFFu + ((u >> 16) & 1u)) >> 16;   // round-to-nearest-even
  return (unsigned short)u;
}
static __device__ __forceinline__ float bf2f(unsigned short h) {
  return __uint_as_float(((unsigned int)h) << 16);
}

// A fragment: 16x32 bf16, M = lane&15, K split 0..7/16..23 (lanes<16) vs
// 8..15/24..31 (lanes>=16), pair-packed per VGPR (ISA 7.12.2).
static __device__ __forceinline__ v16bf frag_a(const unsigned short* p, int ld, int lane) {
  const int m = lane & 15, hf = lane >> 4;
  const unsigned short* r = p + m * ld;
  v8u u;
#pragma unroll
  for (int v = 0; v < 8; ++v) {
    int k = 2 * (((v & 4) << 1) + hf * 4 + (v & 3));
    u[v] = *reinterpret_cast<const unsigned int*>(r + k);
  }
  return __builtin_bit_cast(v16bf, u);
}

// B fragment: 32x16 bf16 read from N-major memory (mem[n][k]), N = lane&15,
// K split 0..7/16..23 (lanes<16) vs 8..15/24..31 (lanes>=16), pair-packed per VGPR.
static __device__ __forceinline__ v16bf frag_b(const unsigned short* p, int ld, int lane) {
  const int n = lane & 15, hf = lane >> 4;
  const unsigned short* r = p + n * ld + hf * 8;
  v8u u;
#pragma unroll
  for (int v = 0; v < 8; ++v)
    u[v] = *reinterpret_cast<const unsigned int*>(r + 2 * (((v & 4) << 1) + (v & 3)));
  return __builtin_bit_cast(v16bf, u);
}

static __device__ __forceinline__ v8f wmma_bf16(v16bf a, v16bf b, v8f c) {
  return __builtin_amdgcn_wmma_f32_16x16x32_bf16(false, a, false, b, (short)0, c,
                                                 false, false);
}

// async 16-byte global -> LDS copy (per-lane); tracked by ASYNCcnt
static __device__ __forceinline__ void async_g2l_b128(unsigned lds_addr,
                                                      const void* gptr) {
  asm volatile("global_load_async_to_lds_b128 %0, %1, off"
               :: "v"(lds_addr), "v"(gptr) : "memory");
}
static __device__ __forceinline__ void wait_async0() {
  asm volatile("s_wait_asynccnt 0x0" ::: "memory");
}

// ---------------------------------------------------------------------------
// elementwise conversion kernels
// ---------------------------------------------------------------------------
__global__ void k_cvt_bf16(const float* __restrict__ src,
                           unsigned short* __restrict__ dst, int n) {
  int i = blockIdx.x * blockDim.x + threadIdx.x;
  if (i < n) dst[i] = f2bf(src[i]);
}

// W[R,C] fp32 -> Wt[C,R] bf16 (N-major, so it can be a WMMA B operand)
__global__ void k_cvt_transpose(const float* __restrict__ src,
                                unsigned short* __restrict__ dst, int R, int C) {
  int i = blockIdx.x * blockDim.x + threadIdx.x;
  if (i < R * C) {
    int r = i / C, c = i % C;
    dst[c * R + r] = f2bf(src[i]);
  }
}

__global__ void k_gate(const float* __restrict__ vraw, const float* __restrict__ graw,
                       const float* __restrict__ bg, unsigned short* __restrict__ out,
                       int n, int cols) {
  int i = blockIdx.x * blockDim.x + threadIdx.x;
  if (i < n) {
    float g = graw[i] + bg[i % cols];
    float s = 1.0f / (1.0f + __expf(-g));
    out[i] = f2bf(vraw[i] * 2.0f * s);
  }
}

// ---------------------------------------------------------------------------
// GEMM: C[4096 x 512] = A_bf16 @ Bt_bf16 (Bt is N-major).  8 waves/WG, each
// wave owns a 16x64 strip.  The shared 64(n) x 32(k) B tile is staged into
// LDS with async b128 copies (one per thread), double-buffered: the issue of
// tile k+1 immediately follows the barrier so the DMA overlaps the 4 WMMAs
// and A-fragment loads of tile k.
// mode 0: phi epilogue (elu+1) -> bf16 out;  mode 1: raw f32 out.
// ---------------------------------------------------------------------------
__global__ __launch_bounds__(256) void k_gemm(const unsigned short* __restrict__ A,
                                              const unsigned short* __restrict__ Bt,
                                              float* __restrict__ outF,
                                              unsigned short* __restrict__ outB,
                                              int Kdim, int mode) {
  __shared__ unsigned short Bs[2][64][32];
  const int tid = threadIdx.x;
  const int lane = tid & 31, w = tid >> 5;
  const int m0 = blockIdx.y * 128 + w * 16;
  const int n0 = blockIdx.x * 64;

  const int sn = tid >> 2, sc = (tid & 3) * 8;   // this thread's 16B of the tile
  auto stage = [&](int k0, int buf) {
    unsigned lds = (unsigned)(size_t)(&Bs[buf][sn][sc]);
    async_g2l_b128(lds, Bt + (size_t)(n0 + sn) * Msz + k0 + sc);
  };

  v8f acc[4] = {};
  stage(0, 0);
  const int nks = Kdim / 32;
  for (int ks = 0; ks < nks; ++ks) {
    const int k0 = ks * 32, p = ks & 1;
    wait_async0();        // this wave's async issues have landed in LDS
    __syncthreads();      // everyone's have; prior reads of buffer p done
    if (ks + 1 < nks) stage(k0 + 32, p ^ 1);   // overlap DMA with compute
    __builtin_prefetch(A + (size_t)m0 * Msz + k0 + 64, 0, 1);
    v16bf a = frag_a(A + (size_t)m0 * Msz + k0, Msz, lane);
#pragma unroll
    for (int j = 0; j < 4; ++j) {
      v16bf b = frag_b(&Bs[p][j * 16][0], 32, lane);
      acc[j] = wmma_bf16(a, b, acc[j]);
    }
  }

  const int hf = lane >> 4, nn = lane & 15;
#pragma unroll
  for (int j = 0; j < 4; ++j)
#pragma unroll
    for (int r = 0; r < 8; ++r) {
      int m = m0 + r + hf * 8;
      int n = n0 + j * 16 + nn;
      float val = acc[j][r];
      if (mode == 0) {
        val = val > 0.0f ? val + 1.0f : __expf(val);     // phi = elu(x)+1
        outB[(size_t)m * Msz + n] = f2bf(val);
      } else {
        outF[(size_t)m * Msz + n] = val;
      }
    }
}

// ---------------------------------------------------------------------------
// Chunked causal linear attention.  One WG per (b,h); 16 chunks of 64 rows.
// Carries S (64x64, kv state) in WMMA accumulators, mirrored to LDS in
// B-operand ([v][k]) layout, and k_cum in LDS.  Also emits
// denom[b,h,t] = qf . k_cum + EPS, reused by the att_map kernel.
// ---------------------------------------------------------------------------
__global__ __launch_bounds__(256) void k_linattn(const unsigned short* __restrict__ qf,
                                                 const unsigned short* __restrict__ kf,
                                                 const unsigned short* __restrict__ vg,
                                                 unsigned short* __restrict__ obf,
                                                 float* __restrict__ denomO) {
  __shared__ float          sA[64][65];    // masked intra scores (f32, rowsums)
  __shared__ unsigned short sAbf[64][64];  // same, bf16 A-operand
  __shared__ unsigned short kT[64][64];    // kf chunk transposed: [k][s]
  __shared__ unsigned short vT[64][64];    // vg chunk transposed: [v][s]
  __shared__ unsigned short Sbf[64][64];   // state, B-operand layout [v][k]
  __shared__ float          kcum[64];
  __shared__ float          sinv[64];

  const int tid = threadIdx.x, lane = tid & 31, w = tid >> 5;
  const int bh = blockIdx.x, b = bh >> 3, hh = bh & 7;
  const int col = hh * 64;
  const int st0 = w * 2;

  for (int i = tid; i < 64 * 64; i += 256) (&Sbf[0][0])[i] = 0;
  if (tid < 64) kcum[tid] = 0.0f;
  v8f Sacc[2] = {};
  __syncthreads();

  for (int c = 0; c < 16; ++c) {
    const int rowb = b * Tsz + c * 64;
    // stage kf/vg chunk transposed into LDS
    for (int i = tid; i < 4096; i += 256) {
      int s = i >> 6, d = i & 63;
      kT[d][s] = kf[(size_t)(rowb + s) * Msz + col + d];
      vT[d][s] = vg[(size_t)(rowb + s) * Msz + col + d];
    }
    __syncthreads();

    // (1) intra-chunk causal scores: qf @ kf^T, mask s>t
#pragma unroll
    for (int q = 0; q < 2; ++q) {
      int st = st0 + q, tr = st >> 2, scx = st & 3;
      v8f acc = {};
#pragma unroll
      for (int kk = 0; kk < 2; ++kk) {
        v16bf a = frag_a(qf + (size_t)(rowb + tr * 16) * Msz + col + kk * 32, Msz, lane);
        v16bf bb = frag_b(kf + (size_t)(rowb + scx * 16) * Msz + col + kk * 32, Msz, lane);
        acc = wmma_bf16(a, bb, acc);
      }
      int nn = lane & 15, hf = lane >> 4;
#pragma unroll
      for (int r = 0; r < 8; ++r) {
        int tl = tr * 16 + r + hf * 8, sl = scx * 16 + nn;
        float val = (sl <= tl) ? acc[r] : 0.0f;
        sA[tl][sl]   = val;
        sAbf[tl][sl] = f2bf(val);
      }
    }
    __syncthreads();

    // (2) denom[t] = rowsum(masked scores) + qf[t].k_cum_prev + EPS
    if (tid < 64) {
      float rs = 0.0f;
      for (int s = 0; s < 64; ++s) rs += sA[tid][s];
      float dot = 0.0f;
      const unsigned short* qr = qf + (size_t)(rowb + tid) * Msz + col;
      for (int k = 0; k < 64; ++k) dot += bf2f(qr[k]) * kcum[k];
      float d = rs + dot + EPSv;
      denomO[bh * Tsz + c * 64 + tid] = d;
      sinv[tid] = 1.0f / d;
    }
    __syncthreads();
    // k_cum += column sums of kf chunk (after all k_cum reads)
    if (tid < 64) {
      float cs = 0.0f;
      for (int s = 0; s < 64; ++s) cs += bf2f(kT[tid][s]);
      kcum[tid] += cs;
    }

    // (3) numer = masked_scores @ vg  +  qf @ S_prev ; divide by denom
#pragma unroll
    for (int q = 0; q < 2; ++q) {
      int st = st0 + q, tr = st >> 2, vc = st & 3;
      v8f acc = {};
#pragma unroll
      for (int s0 = 0; s0 < 2; ++s0) {
        v16bf a = frag_a(&sAbf[tr * 16][s0 * 32], 64, lane);
        v16bf bb = frag_b(&vT[vc * 16][s0 * 32], 64, lane);
        acc = wmma_bf16(a, bb, acc);
      }
#pragma unroll
      for (int kk = 0; kk < 2; ++kk) {
        v16bf a = frag_a(qf + (size_t)(rowb + tr * 16) * Msz + col + kk * 32, Msz, lane);
        v16bf bb = frag_b(&Sbf[vc * 16][kk * 32], 64, lane);
        acc = wmma_bf16(a, bb, acc);
      }
      int nn = lane & 15, hf = lane >> 4;
#pragma unroll
      for (int r = 0; r < 8; ++r) {
        int tl = tr * 16 + r + hf * 8, vv = vc * 16 + nn;
        obf[(size_t)(rowb + tl) * Msz + col + vv] = f2bf(acc[r] * sinv[tl]);
      }
    }
    __syncthreads();   // all S_prev reads done before state update

    // (4) S += kf^T @ vg  (accumulators persist; mirror to Sbf [v][k])
#pragma unroll
    for (int q = 0; q < 2; ++q) {
      int st = st0 + q, kr = st >> 2, vc = st & 3;
#pragma unroll
      for (int s0 = 0; s0 < 2; ++s0) {
        v16bf a = frag_a(&kT[kr * 16][s0 * 32], 64, lane);
        v16bf bb = frag_b(&vT[vc * 16][s0 * 32], 64, lane);
        Sacc[q] = wmma_bf16(a, bb, Sacc[q]);
      }
      int nn = lane & 15, hf = lane >> 4;
#pragma unroll
      for (int r = 0; r < 8; ++r) {
        int km = kr * 16 + r + hf * 8, vv = vc * 16 + nn;
        Sbf[vv][km] = f2bf(Sacc[q][r]);
      }
    }
    __syncthreads();
  }
}

// ---------------------------------------------------------------------------
// att_map: one WG per (b,h, 64-row block).  Lower-triangle tiles via WMMA,
// scaled by 1/denom; strict upper triangle written as zeros (single pass).
// ---------------------------------------------------------------------------
__global__ __launch_bounds__(256) void k_att(const unsigned short* __restrict__ qf,
                                             const unsigned short* __restrict__ kf,
                                             const float* __restrict__ denom,
                                             float* __restrict__ att) {
  __shared__ float sinv[64];
  const int tid = threadIdx.x, lane = tid & 31, w = tid >> 5;
  const int tb = blockIdx.x;            // 0..15 row block
  const int bh = blockIdx.y;            // 0..31
  const int b = bh >> 3, hh = bh & 7;
  const int col = hh * 64, t0 = tb * 64;
  if (tid < 64) sinv[tid] = 1.0f / denom[bh * Tsz + t0 + tid];
  __syncthreads();

  float* attBH = att + (size_t)bh * Tsz * Tsz;
  const int st0 = w * 2;
  for (int sb = 0; sb <= tb; ++sb) {
#pragma unroll
    for (int q = 0; q < 2; ++q) {
      int st = st0 + q, tr = st >> 2, sc = st & 3;
      v8f acc = {};
#pragma unroll
      for (int kk = 0; kk < 2; ++kk) {
        v16bf a = frag_a(qf + (size_t)(b * Tsz + t0 + tr * 16) * Msz + col + kk * 32,
                         Msz, lane);
        v16bf bb = frag_b(kf + (size_t)(b * Tsz + sb * 64 + sc * 16) * Msz + col + kk * 32,
                          Msz, lane);
        acc = wmma_bf16(a, bb, acc);
      }
      int nn = lane & 15, hf = lane >> 4;
#pragma unroll
      for (int r = 0; r < 8; ++r) {
        int tl = tr * 16 + r + hf * 8;
        int tg = t0 + tl;
        int sg = sb * 64 + sc * 16 + nn;
        float val = acc[r];
        if (sb == tb && sg > tg) val = 0.0f;
        attBH[(size_t)tg * Tsz + sg] = val * sinv[tl];
      }
    }
  }
  // zero-fill strict upper-triangle column range
  int sstart = (tb + 1) * 64;
  int cols = Tsz - sstart;
  if (cols > 0) {
    int cq = cols >> 2;
    for (int i = tid; i < 64 * cq; i += 256) {
      int r = i / cq, cc = (i % cq) << 2;
      float4 z = {0.0f, 0.0f, 0.0f, 0.0f};
      *reinterpret_cast<float4*>(&attBH[(size_t)(t0 + r) * Tsz + sstart + cc]) = z;
    }
  }
}

// ---------------------------------------------------------------------------
// host launcher
// ---------------------------------------------------------------------------
extern "C" void kernel_launch(void* const* d_in, const int* in_sizes, int n_in,
                              void* d_out, int out_size, void* d_ws, size_t ws_size,
                              hipStream_t stream) {
  const float* query = (const float*)d_in[0];
  const float* key   = (const float*)d_in[1];
  const float* value = (const float*)d_in[2];
  const float* Wq    = (const float*)d_in[3];
  const float* Wk    = (const float*)d_in[4];
  const float* Wv    = (const float*)d_in[5];
  const float* Wg    = (const float*)d_in[6];
  const float* bg    = (const float*)d_in[7];
  const float* Wo    = (const float*)d_in[8];

  const size_t NBT = (size_t)Bsz * Tsz * Msz;  // 2,097,152 elements
  const size_t NW  = (size_t)Msz * HK;         //   262,144 elements

  char* ws = (char*)d_ws;
  size_t off = 0;
  auto take = [&](size_t bytes) { char* p = ws + off; off = (off + bytes + 255) & ~(size_t)255; return p; };

  unsigned short* qbf   = (unsigned short*)take(NBT * 2);
  unsigned short* kbf   = (unsigned short*)take(NBT * 2);
  unsigned short* vbf   = (unsigned short*)take(NBT * 2);
  unsigned short* WqT   = (unsigned short*)take(NW * 2);
  unsigned short* WkT   = (unsigned short*)take(NW * 2);
  unsigned short* WvT   = (unsigned short*)take(NW * 2);
  unsigned short* WgT   = (unsigned short*)take(NW * 2);
  unsigned short* WoT   = (unsigned short*)take(NW * 2);
  unsigned short* qfb   = (unsigned short*)take(NBT * 2);
  unsigned short* kfb   = (unsigned short*)take(NBT * 2);
  unsigned short* vgb   = (unsigned short*)take(NBT * 2);
  float*          vraw  = (float*)take(NBT * 4);
  float*          graw  = (float*)take(NBT * 4);
  unsigned short* obf   = (unsigned short*)take(NBT * 2);
  float*          denomB= (float*)take((size_t)Bsz * Hsz * Tsz * 4);

  float* out_main = (float*)d_out;                  // [B,T,M]
  float* att_map  = out_main + NBT;                 // [B,H,T,T]

  // 1) fp32 -> bf16 activations, transposed bf16 weights
  {
    int n = (int)NBT;
    dim3 g((n + 255) / 256), blk(256);
    k_cvt_bf16<<<g, blk, 0, stream>>>(query, qbf, n);
    k_cvt_bf16<<<g, blk, 0, stream>>>(key,   kbf, n);
    k_cvt_bf16<<<g, blk, 0, stream>>>(value, vbf, n);
    int nw = (int)NW;
    dim3 gw((nw + 255) / 256);
    k_cvt_transpose<<<gw, blk, 0, stream>>>(Wq, WqT, Msz, HK);
    k_cvt_transpose<<<gw, blk, 0, stream>>>(Wk, WkT, Msz, HK);
    k_cvt_transpose<<<gw, blk, 0, stream>>>(Wv, WvT, Msz, HK);
    k_cvt_transpose<<<gw, blk, 0, stream>>>(Wg, WgT, Msz, HK);
    k_cvt_transpose<<<gw, blk, 0, stream>>>(Wo, WoT, HK, Msz);
  }

  // 2) projections (phi fused into q/k epilogue)
  {
    dim3 g(HK / 64, (Bsz * Tsz) / 128), blk(256);
    k_gemm<<<g, blk, 0, stream>>>(qbf, WqT, nullptr, qfb, Msz, 0);  // qf = phi(q)
    k_gemm<<<g, blk, 0, stream>>>(kbf, WkT, nullptr, kfb, Msz, 0);  // kf = phi(k)
    k_gemm<<<g, blk, 0, stream>>>(vbf, WvT, vraw, nullptr, Msz, 1); // v raw
    k_gemm<<<g, blk, 0, stream>>>(vbf, WgT, graw, nullptr, Msz, 1); // gate raw
  }

  // 3) value gating: vg = v * 2*sigmoid(g + bg)
  {
    int n = (int)NBT;
    k_gate<<<dim3((n + 255) / 256), dim3(256), 0, stream>>>(vraw, graw, bg, vgb, n, HK);
  }

  // 4) chunked causal linear attention -> o (bf16) and denom
  k_linattn<<<dim3(Bsz * Hsz), dim3(256), 0, stream>>>(qfb, kfb, vgb, obf, denomB);

  // 5) att_map (reuses denom as the row normalizer)
  k_att<<<dim3(Tsz / 64, Bsz * Hsz), dim3(256), 0, stream>>>(qfb, kfb, denomB, att_map);

  // 6) out = o @ Wo
  {
    dim3 g(Msz / 64, (Bsz * Tsz) / 128), blk(256);
    k_gemm<<<g, blk, 0, stream>>>(obf, WoT, out_main, nullptr, HK, 1);
  }
}
